// CausalSelfAttention_37288906064117
// MI455X (gfx1250) — matrix operand order repaired and checked by reading, not compiled, not run on hardware
//
#include <hip/hip_runtime.h>
#include <hip/hip_bf16.h>

// Problem constants (match reference)
#define BB 2
#define TT 2048
#define CC 1024
#define HH 16
#define DD 64           // head dim
#define ROPE_BASE 10000.0f

typedef __attribute__((ext_vector_type(8)))  float  v8f;
typedef __attribute__((ext_vector_type(8)))  __bf16 v8bf;
typedef __attribute__((ext_vector_type(16))) __bf16 v16bf;

// ---------------- helpers ----------------

static __device__ inline __bf16 f2bf(float f) {
    union { float f; unsigned u; } v; v.f = f;
    unsigned r = v.u + 0x7FFFu + ((v.u >> 16) & 1u);   // round-to-nearest-even
    union { unsigned short s; __bf16 b; } o;
    o.s = (unsigned short)(r >> 16);
    return o.b;
}

static __device__ inline v8f wmma_bf16(v16bf a, v16bf b, v8f c) {
    // D = A(16x32) * B(32x16) + C(16x16), f32 accumulate
    return __builtin_amdgcn_wmma_f32_16x16x32_bf16(false, a, false, b, (short)0, c, false, false);
}

// A-fragment (16-bit, 16x32): lane holds row (l&15); elements 0..15 at K=kbase+0..15,
// where kbase already includes half*16.
static __device__ inline v16bf load_a16(const __bf16* rowp, int kbase) {
    v8bf lo = *reinterpret_cast<const v8bf*>(rowp + kbase);
    v8bf hi = *reinterpret_cast<const v8bf*>(rowp + kbase + 8);
    v16bf r;
#pragma unroll
    for (int i = 0; i < 8; ++i) { r[i] = lo[i]; r[i + 8] = hi[i]; }
    return r;
}

// B-fragment (16-bit, 32x16): lane holds col (l&15); 16 contiguous contraction elements.
static __device__ inline v16bf load_b16(const __bf16* p) {
    v8bf lo = *reinterpret_cast<const v8bf*>(p);
    v8bf hi = *reinterpret_cast<const v8bf*>(p + 8);
    v16bf r;
#pragma unroll
    for (int i = 0; i < 8; ++i) { r[i] = lo[i]; r[i + 8] = hi[i]; }
    return r;
}

// ---------------- kernel 1: f32 -> bf16 ----------------

__global__ void cvt_f32_bf16(const float* __restrict__ in, __bf16* __restrict__ out, int n) {
    int i = blockIdx.x * blockDim.x + threadIdx.x;
    if (i < n) out[i] = f2bf(in[i]);
}

// ---------------- kernel 2: GEMM  C[M,N] = A[M,K] * B[N,K]^T + bias ----------------
// One wave computes a 64x64 tile (4x4 WMMA register blocking: 16 WMMAs per k-step
// against 8 fragment loads -> 4x less L2 fragment traffic than 16x16 tiles).
// A,B bf16 row-major (K contiguous), C f32.  M % 64 == 0, N % 64 == 0 assumed.

__global__ __launch_bounds__(128) void gemm_bf16(const __bf16* __restrict__ A,
                                                 const __bf16* __restrict__ Bw,
                                                 const float* __restrict__ bias,
                                                 float* __restrict__ Cout,
                                                 int N, int Kd, int tiles_n, int tiles_total) {
    const int wid = blockIdx.x * 4 + (threadIdx.x >> 5);
    if (wid >= tiles_total) return;        // whole-wave uniform -> EXEC stays all-ones
    const int lane = threadIdx.x & 31;
    const int half = lane >> 4;
    const int l15  = lane & 15;
    const int m0 = (wid / tiles_n) * 64;
    const int n0 = (wid % tiles_n) * 64;

    const __bf16* arow[4];
    const __bf16* brow[4];
#pragma unroll
    for (int i = 0; i < 4; ++i) {
        arow[i] = A  + (size_t)(m0 + i * 16 + l15) * Kd;
        brow[i] = Bw + (size_t)(n0 + i * 16 + l15) * Kd + half * 16;
    }

    v8f acc[4][4];
#pragma unroll
    for (int i = 0; i < 4; ++i)
#pragma unroll
        for (int j = 0; j < 4; ++j) acc[i][j] = (v8f){};

    for (int k = 0; k < Kd; k += 32) {
        v16bf af[4], bfr[4];
#pragma unroll
        for (int i = 0; i < 4; ++i) af[i]  = load_a16(arow[i], k + half * 16);
#pragma unroll
        for (int j = 0; j < 4; ++j) bfr[j] = load_b16(brow[j] + k);
#pragma unroll
        for (int i = 0; i < 4; ++i)
#pragma unroll
            for (int j = 0; j < 4; ++j)
                acc[i][j] = wmma_bf16(af[i], bfr[j], acc[i][j]);
    }

#pragma unroll
    for (int i = 0; i < 4; ++i) {
#pragma unroll
        for (int j = 0; j < 4; ++j) {
            const float bz = bias[n0 + j * 16 + l15];
#pragma unroll
            for (int r = 0; r < 8; ++r) {
                const int row = m0 + i * 16 + r + half * 8;   // C-layout: M = r + 8*(lane>=16)
                Cout[(size_t)row * N + n0 + j * 16 + l15] = acc[i][j][r] + bz;
            }
        }
    }
}

// ---------------- kernel 3: RoPE + head reshape ----------------
// qkv f32 (B,T,3C) -> Q,K bf16 (B,H,T,D) (rope'd), V bf16 transposed (B,H,D,T)

__global__ void rope_reshape(const float* __restrict__ qkv,
                             __bf16* __restrict__ Q, __bf16* __restrict__ K,
                             __bf16* __restrict__ Vt) {
    int idx = blockIdx.x * blockDim.x + threadIdx.x;         // over B*T*(C/2) pairs
    const int total = BB * TT * (CC / 2);
    if (idx >= total) return;
    const int cpair = idx % (CC / 2);
    const int bt    = idx / (CC / 2);
    const int t = bt % TT, b = bt / TT;
    const int c = cpair * 2;
    const int h = c / DD, d = c % DD;                        // d even

    const float theta = __expf(-__logf(ROPE_BASE) * (float)d / (float)DD);
    const float ang = (float)t * theta;
    const float cs = cosf(ang), sn = sinf(ang);

    const float* base = qkv + (size_t)(b * TT + t) * (3 * CC);
    const float q0 = base[c],          q1 = base[c + 1];
    const float k0 = base[CC + c],     k1 = base[CC + c + 1];
    const float v0 = base[2 * CC + c], v1 = base[2 * CC + c + 1];

    const size_t qk = ((size_t)(b * HH + h) * TT + t) * DD + d;
    Q[qk]     = f2bf(q0 * cs - q1 * sn);
    Q[qk + 1] = f2bf(q1 * cs + q0 * sn);
    K[qk]     = f2bf(k0 * cs - k1 * sn);
    K[qk + 1] = f2bf(k1 * cs + k0 * sn);

    const size_t vo = ((size_t)(b * HH + h) * DD + d) * TT + t;
    Vt[vo]      = f2bf(v0);
    Vt[vo + TT] = f2bf(v1);
}

// ---------------- kernel 4: causal flash attention ----------------
// One wave per 16-row q-tile; k-blocks of 32; WMMA for QK^T and PV.
// Y written as (B,T,C) bf16 ready for the projection GEMM.

__global__ __launch_bounds__(128) void attn_fwd(const __bf16* __restrict__ Q,
                                                const __bf16* __restrict__ K,
                                                const __bf16* __restrict__ Vt,
                                                __bf16* __restrict__ Y) {
    __shared__ __bf16 lds_p[4][16][32];    // per-wave P-tile transpose staging (4 KB)

    const int lane = threadIdx.x & 31;
    const int w    = threadIdx.x >> 5;
    const int wid  = blockIdx.x * 4 + w;   // 0 .. B*H*(T/16)-1 = 4095
    const int qt = wid & (TT / 16 - 1);    // T/16 = 128
    const int bh = wid >> 7;
    const int q0 = qt * 16;
    const int half = lane >> 4;
    const int l15  = lane & 15;

    const __bf16* Qb = Q  + (size_t)bh * TT * DD;
    const __bf16* Kb = K  + (size_t)bh * TT * DD;
    const __bf16* Vb = Vt + (size_t)bh * DD * TT;

    // Q A-fragments for contraction d=[0,32) and [32,64)
    const __bf16* qrow = Qb + (size_t)(q0 + l15) * DD;
    const v16bf aq0 = load_a16(qrow, 0  + half * 16);
    const v16bf aq1 = load_a16(qrow, 32 + half * 16);

    v8f acc0 = {}, acc1 = {}, acc2 = {}, acc3 = {};
    float m_i[8], l_i[8];
#pragma unroll
    for (int r = 0; r < 8; ++r) { m_i[r] = -3.0e38f; l_i[r] = 0.0f; }

    for (int k0 = 0; k0 <= q0 + 15; k0 += 32) {
        // ---- scores S (16 x 32) = Q (16x64) * K^T ----
        v8f s0 = {}, s1 = {};
        {
            const __bf16* kr0 = Kb + (size_t)(k0 + l15) * DD;        // cols k0..k0+15
            const __bf16* kr1 = Kb + (size_t)(k0 + 16 + l15) * DD;   // cols k0+16..k0+31
            s0 = wmma_bf16(aq0, load_b16(kr0 + 0  + half * 16), s0);
            s0 = wmma_bf16(aq1, load_b16(kr0 + 32 + half * 16), s0);
            s1 = wmma_bf16(aq0, load_b16(kr1 + 0  + half * 16), s1);
            s1 = wmma_bf16(aq1, load_b16(kr1 + 32 + half * 16), s1);
        }

        // ---- scale + causal mask + online softmax (rows r -> M = r + 8*half) ----
        float pmax[8];
#pragma unroll
        for (int r = 0; r < 8; ++r) {
            const int qrowi = q0 + r + half * 8;
            float a = s0[r] * 0.125f;
            float b = s1[r] * 0.125f;
            if (k0 + l15 > qrowi)      a = -3.0e38f;
            if (k0 + 16 + l15 > qrowi) b = -3.0e38f;
            s0[r] = a; s1[r] = b;
            pmax[r] = fmaxf(a, b);
        }
#pragma unroll
        for (int mk = 1; mk < 16; mk <<= 1) {
#pragma unroll
            for (int r = 0; r < 8; ++r)
                pmax[r] = fmaxf(pmax[r], __shfl_xor(pmax[r], mk, 32));
        }
        float corr[8], psum[8];
#pragma unroll
        for (int r = 0; r < 8; ++r) {
            const float mnew = fmaxf(m_i[r], pmax[r]);
            corr[r] = __expf(m_i[r] - mnew);
            m_i[r] = mnew;
            const float p0 = __expf(s0[r] - mnew);
            const float p1 = __expf(s1[r] - mnew);
            s0[r] = p0; s1[r] = p1;
            psum[r] = p0 + p1;
        }
#pragma unroll
        for (int mk = 1; mk < 16; mk <<= 1) {
#pragma unroll
            for (int r = 0; r < 8; ++r)
                psum[r] += __shfl_xor(psum[r], mk, 32);
        }
#pragma unroll
        for (int r = 0; r < 8; ++r) {
            l_i[r] = l_i[r] * corr[r] + psum[r];
            acc0[r] *= corr[r]; acc1[r] *= corr[r];
            acc2[r] *= corr[r]; acc3[r] *= corr[r];
        }

        // ---- C-layout P -> LDS -> A-layout bf16 fragment ----
#pragma unroll
        for (int r = 0; r < 8; ++r) {
            const int row = r + half * 8;
            lds_p[w][row][l15]      = f2bf(s0[r]);
            lds_p[w][row][16 + l15] = f2bf(s1[r]);
        }
        asm volatile("s_wait_dscnt 0x0" ::: "memory");   // gfx1250 split DS counter wait
        const v16bf pa = load_a16(&lds_p[w][l15][0], half * 16);

        // ---- Y (16x64) += P (16x32) * V (32x64), V stored (D,T) ----
        const size_t vofs = (size_t)k0 + half * 16;
        acc0 = wmma_bf16(pa, load_b16(Vb + (size_t)(0  + l15) * TT + vofs), acc0);
        acc1 = wmma_bf16(pa, load_b16(Vb + (size_t)(16 + l15) * TT + vofs), acc1);
        acc2 = wmma_bf16(pa, load_b16(Vb + (size_t)(32 + l15) * TT + vofs), acc2);
        acc3 = wmma_bf16(pa, load_b16(Vb + (size_t)(48 + l15) * TT + vofs), acc3);
    }

    // ---- normalize + write Y as (B,T,C) bf16 ----
    const int b = bh >> 4;      // / H
    const int h = bh & 15;
#pragma unroll
    for (int r = 0; r < 8; ++r) {
        const float inv = 1.0f / l_i[r];
        const int tpos = q0 + r + half * 8;
        const size_t rowo = (size_t)(b * TT + tpos) * CC + h * DD;
        Y[rowo + 0  + l15] = f2bf(acc0[r] * inv);
        Y[rowo + 16 + l15] = f2bf(acc1[r] * inv);
        Y[rowo + 32 + l15] = f2bf(acc2[r] * inv);
        Y[rowo + 48 + l15] = f2bf(acc3[r] * inv);
    }
}

// ---------------- host orchestration ----------------

extern "C" void kernel_launch(void* const* d_in, const int* in_sizes, int n_in,
                              void* d_out, int out_size, void* d_ws, size_t ws_size,
                              hipStream_t stream) {
    (void)in_sizes; (void)n_in; (void)out_size; (void)ws_size;
    const float* x     = (const float*)d_in[0];
    const float* Wqkv  = (const float*)d_in[1];
    const float* bqkv  = (const float*)d_in[2];
    const float* Wproj = (const float*)d_in[3];
    const float* bproj = (const float*)d_in[4];
    float* out = (float*)d_out;

    char* ws = (char*)d_ws;
    size_t off = 0;
    auto carve = [&](size_t bytes) -> void* {
        void* p = ws + off;
        off += (bytes + 255) & ~(size_t)255;
        return p;
    };
    const size_t nX   = (size_t)BB * TT * CC;       // 4,194,304
    const size_t nWq  = (size_t)3 * CC * CC;        // 3,145,728
    const size_t nWp  = (size_t)CC * CC;            // 1,048,576
    const size_t nQKV = (size_t)BB * TT * 3 * CC;   // 12,582,912
    const size_t nHd  = (size_t)BB * HH * TT * DD;  // 4,194,304

    __bf16* Xbf   = (__bf16*)carve(nX * 2);
    __bf16* Wqbf  = (__bf16*)carve(nWq * 2);
    __bf16* Wpbf  = (__bf16*)carve(nWp * 2);
    float*  qkv   = (float*)carve(nQKV * 4);
    __bf16* Qb    = (__bf16*)carve(nHd * 2);
    __bf16* Kb    = (__bf16*)carve(nHd * 2);
    __bf16* Vtb   = (__bf16*)carve(nHd * 2);
    __bf16* Ybf   = (__bf16*)carve(nX * 2);

    // 1) convert inputs to bf16
    cvt_f32_bf16<<<(int)((nX  + 255) / 256), 256, 0, stream>>>(x,     Xbf,  (int)nX);
    cvt_f32_bf16<<<(int)((nWq + 255) / 256), 256, 0, stream>>>(Wqkv,  Wqbf, (int)nWq);
    cvt_f32_bf16<<<(int)((nWp + 255) / 256), 256, 0, stream>>>(Wproj, Wpbf, (int)nWp);

    // 2) qkv = x @ Wqkv^T + bqkv   (M=4096, N=3072, K=1024), 64x64 wave tiles
    {
        const int tiles_n = (3 * CC) / 64;                       // 48
        const int tiles   = (BB * TT / 64) * tiles_n;            // 3072
        gemm_bf16<<<(tiles + 3) / 4, 128, 0, stream>>>(Xbf, Wqbf, bqkv, qkv,
                                                       3 * CC, CC, tiles_n, tiles);
    }

    // 3) RoPE + reshape to heads
    {
        const int total = BB * TT * (CC / 2);
        rope_reshape<<<(total + 255) / 256, 256, 0, stream>>>(qkv, Qb, Kb, Vtb);
    }

    // 4) causal flash attention (4096 waves, 4 per block)
    attn_fwd<<<(BB * HH * (TT / 16)) / 4, 128, 0, stream>>>(Qb, Kb, Vtb, Ybf);

    // 5) out = y @ Wproj^T + bproj  (M=4096, N=1024, K=1024), 64x64 wave tiles
    {
        const int tiles_n = CC / 64;                             // 16
        const int tiles   = (BB * TT / 64) * tiles_n;            // 1024
        gemm_bf16<<<(tiles + 3) / 4, 128, 0, stream>>>(Ybf, Wpbf, bproj, out,
                                                       CC, CC, tiles_n, tiles);
    }
}
